// FullAttention_17248588661295
// MI455X (gfx1250) — matrix operand order repaired and checked by reading, not compiled, not run on hardware
//
#include <hip/hip_runtime.h>
#include <hip/hip_bf16.h>

// ---------------------------------------------------------------------------
// Problem constants (from reference)
// ---------------------------------------------------------------------------
#define B_    2
#define S_    2048
#define HID_  512
#define NH_   8
#define NKV_  2
#define HD_   256
#define ROT_  64
// groups = NH_/NKV_ = 4; score scale 1/sqrt(HD)=1/16 folded into Q (exact in bf16)

typedef __bf16 bf16_t;
typedef __attribute__((ext_vector_type(16))) __bf16 v16bf;
typedef __attribute__((ext_vector_type(8)))  __bf16 v8bf;
typedef __attribute__((ext_vector_type(8)))  float  v8f;
typedef __attribute__((ext_vector_type(4)))  int    v4i;

// Async global->LDS path (CDNA5, ASYNCcnt-tracked); probe-verified signature:
// (addrspace(1) v4i*, addrspace(3) v4i*, imm offset, imm cpol).
#if defined(__gfx1250__) &&                                            \
    __has_builtin(__builtin_amdgcn_global_load_async_to_lds_b128) &&   \
    __has_builtin(__builtin_amdgcn_s_wait_asynccnt)
#define USE_ASYNC_LDS 1
typedef __attribute__((address_space(1))) v4i as1_v4i;
typedef __attribute__((address_space(3))) v4i as3_v4i;
#else
#define USE_ASYNC_LDS 0
#endif

// ---------------------------------------------------------------------------
// WMMA helpers (CDNA5 wave32, 16x16x32 bf16 -> f32)
// ---------------------------------------------------------------------------
__device__ __forceinline__ v8f wmma_bf16(v16bf a, v16bf b, v8f c) {
  return __builtin_amdgcn_wmma_f32_16x16x32_bf16(false, a, false, b, (short)0, c,
                                                 false, false);
}

// A-fragment: lane L (row = L%16), K = khalf+0..7 then 16+khalf+0..7,
// khalf = (L/16)*8. Two contiguous 16B loads.
__device__ __forceinline__ v16bf load_frag_a(const bf16_t* p0, const bf16_t* p1) {
  v8bf lo = *reinterpret_cast<const v8bf*>(p0);
  v8bf hi = *reinterpret_cast<const v8bf*>(p1);
  v16bf r;
#pragma unroll
  for (int i = 0; i < 8; ++i) { r[i] = lo[i]; r[i + 8] = hi[i]; }
  return r;
}

// B-fragment: lane L (col n = L%16), K = khalf+0..7 then 16+khalf+0..7, khalf = (L/16)*8.
__device__ __forceinline__ v16bf load_frag_b(const bf16_t* p) {
  v8bf lo = *reinterpret_cast<const v8bf*>(p);
  v8bf hi = *reinterpret_cast<const v8bf*>(p + 16);
  v16bf r;
#pragma unroll
  for (int i = 0; i < 8; ++i) { r[i] = lo[i]; r[i + 8] = hi[i]; }
  return r;
}

// ---------------------------------------------------------------------------
// Elementwise conversion / transposition kernels (bandwidth-trivial)
// ---------------------------------------------------------------------------
__global__ void cvt_f32_bf16_kernel(const float* __restrict__ src,
                                    bf16_t* __restrict__ dst, int n) {
  int i = blockIdx.x * 256 + threadIdx.x;
  if (i < n) dst[i] = (bf16_t)src[i];
}

// dst (N x K, row-major bf16) = transpose(src (K x N, row-major f32))
__global__ void transpose_cvt_kernel(const float* __restrict__ src,
                                     bf16_t* __restrict__ dst, int K, int N) {
  int idx = blockIdx.x * 256 + threadIdx.x;
  if (idx >= K * N) return;
  int n = idx / K, k = idx - n * K;
  dst[idx] = (bf16_t)src[(size_t)k * N + n];
}

// Vt (B,NKV,HD,S) bf16 = transpose of V (B,S,NKV,HD) f32
__global__ void v_transpose_kernel(const float* __restrict__ V,
                                   bf16_t* __restrict__ Vt) {
  int idx = blockIdx.x * 256 + threadIdx.x;
  int total = B_ * NKV_ * HD_ * S_;
  if (idx >= total) return;
  int s = idx % S_;
  int r = idx / S_;
  int hd = r % HD_; r /= HD_;
  int kv = r % NKV_;
  int b  = r / NKV_;
  Vt[idx] = (bf16_t)V[(((size_t)b * S_ + s) * NKV_ + kv) * HD_ + hd];
}

// ---------------------------------------------------------------------------
// Fused per-head RMS-norm + partial RoPE; wave per row of 256, output bf16.
// out_scale folds the attention score scale into Q (1/16: exact in bf16).
// ---------------------------------------------------------------------------
__global__ __launch_bounds__(256) void rmsnorm_rope_kernel(
    const float* __restrict__ X, bf16_t* __restrict__ Y,
    const float* __restrict__ w, const float* __restrict__ cosb,
    const float* __restrict__ sinb, int nheads, float out_scale) {
  int lane = threadIdx.x & 31;
  int wave = threadIdx.x >> 5;
  int row = blockIdx.x * 8 + wave;
  if (row >= B_ * S_ * nheads) return;
  int s = (row / nheads) % S_;
  const float* x = X + (size_t)row * HD_;
  float v[8];
  float ss = 0.0f;
#pragma unroll
  for (int j = 0; j < 8; ++j) { v[j] = x[lane * 8 + j]; ss += v[j] * v[j]; }
#pragma unroll
  for (int off = 1; off < 32; off <<= 1) ss += __shfl_xor(ss, off, 32);
  float inv = 1.0f / sqrtf(ss * (1.0f / HD_) + 1e-6f);
#pragma unroll
  for (int j = 0; j < 8; ++j) v[j] = v[j] * inv * (1.0f + w[lane * 8 + j]);
  if (lane < ROT_ / 8) {
#pragma unroll
    for (int t = 0; t < 4; ++t) {
      int pair = lane * 4 + t;
      float c = cosb[s * (ROT_ / 2) + pair];
      float sn = sinb[s * (ROT_ / 2) + pair];
      float a = v[t * 2], b = v[t * 2 + 1];
      v[t * 2]     = a * c - b * sn;
      v[t * 2 + 1] = a * sn + b * c;
    }
  }
  bf16_t* y = Y + (size_t)row * HD_;
#pragma unroll
  for (int j = 0; j < 8; ++j) y[lane * 8 + j] = (bf16_t)(v[j] * out_scale);
}

// ---------------------------------------------------------------------------
// bf16 WMMA GEMM: C (MxN, f32) = A (MxK, bf16) * Bt^T (Bt is NxK row-major).
// Wave computes 32x64.  k-loop unrolled by 2 with two *named* fragment sets
// (phase A @ k0, phase B @ k0+32) so the double buffer alternates by name:
// no v_mov rotation, loads of one phase overlap WMMAs of the other.
// Requires K % 64 == 0 (holds: 512, 2048).
// ---------------------------------------------------------------------------
__global__ __launch_bounds__(256) void gemm_bf16_kernel(
    const bf16_t* __restrict__ A, const bf16_t* __restrict__ Bt,
    float* __restrict__ C, int M, int N, int K) {
  int lane = threadIdx.x & 31;
  int wave = threadIdx.x >> 5;
  int gw = blockIdx.x * 8 + wave;
  int ntn = N >> 6;
  int tm = gw / ntn;
  int tn = gw - tm * ntn;
  if (tm >= (M >> 5)) return;
  int m0 = tm << 5, n0 = tn << 6;
  int half = lane >> 4, l16 = lane & 15;

  const bf16_t* Arow0 = A + (size_t)(m0 + l16) * K + half * 8;
  const bf16_t* Arow1 = Arow0 + (size_t)16 * K;
  const bf16_t* Brow[4];
#pragma unroll
  for (int t = 0; t < 4; ++t)
    Brow[t] = Bt + (size_t)(n0 + t * 16 + l16) * K + half * 8;

  v8f acc[8] = {};
  // phase-A fragments @ k=0
  v16bf a0A = load_frag_a(Arow0, Arow0 + 16);
  v16bf a1A = load_frag_a(Arow1, Arow1 + 16);
  v16bf bA[4];
#pragma unroll
  for (int t = 0; t < 4; ++t) bA[t] = load_frag_b(Brow[t]);

  for (int k0 = 0; k0 < K; k0 += 64) {
    // stage phase-B fragments @ k0+32 (always in range)
    __builtin_prefetch(Arow0 + k0 + 256, 0, 0);  // global_prefetch_b8
    v16bf a0B = load_frag_a(Arow0 + k0 + 32, Arow0 + k0 + 48);
    v16bf a1B = load_frag_a(Arow1 + k0 + 32, Arow1 + k0 + 48);
    v16bf bB[4];
#pragma unroll
    for (int t = 0; t < 4; ++t) bB[t] = load_frag_b(Brow[t] + k0 + 32);
    // compute phase A (k0)
#pragma unroll
    for (int t = 0; t < 4; ++t) {
      acc[t]     = wmma_bf16(a0A, bA[t], acc[t]);
      acc[4 + t] = wmma_bf16(a1A, bA[t], acc[4 + t]);
    }
    // stage next phase-A fragments @ k0+64
    if (k0 + 64 < K) {
      a0A = load_frag_a(Arow0 + k0 + 64, Arow0 + k0 + 80);
      a1A = load_frag_a(Arow1 + k0 + 64, Arow1 + k0 + 80);
#pragma unroll
      for (int t = 0; t < 4; ++t) bA[t] = load_frag_b(Brow[t] + k0 + 64);
    }
    // compute phase B (k0+32)
#pragma unroll
    for (int t = 0; t < 4; ++t) {
      acc[t]     = wmma_bf16(a0B, bB[t], acc[t]);
      acc[4 + t] = wmma_bf16(a1B, bB[t], acc[4 + t]);
    }
  }
#pragma unroll
  for (int g = 0; g < 2; ++g)
#pragma unroll
    for (int t = 0; t < 4; ++t)
#pragma unroll
      for (int i = 0; i < 8; ++i) {
        int r = g * 16 + i + half * 8;
        C[(size_t)(m0 + r) * N + n0 + t * 16 + l16] = acc[g * 4 + t][i];
      }
}

// ---------------------------------------------------------------------------
// Flash attention (causal, GQA). Wave = 16 query rows x full HD=256.
// K chunks (32 keys x 32 dims bf16) are async-prefetched into a per-wave
// LDS double buffer one chunk ahead (ASYNCcnt), hiding HBM latency behind
// the WMMAs. No block barriers: per-wave trip counts diverge (causal).
// ---------------------------------------------------------------------------
__global__ __launch_bounds__(256) void attn_kernel(
    const bf16_t* __restrict__ Qb, const bf16_t* __restrict__ Kb,
    const bf16_t* __restrict__ Vt, bf16_t* __restrict__ Ob) {
  __shared__ bf16_t plds[8][16 * 32];        // P tile, 1KB/wave
#if USE_ASYNC_LDS
  __shared__ bf16_t klds[8][2][32 * 32];     // K chunk double buffer, 4KB/wave
#endif
  int lane = threadIdx.x & 31;
  int wave = threadIdx.x >> 5;
  int gw = blockIdx.x * 8 + wave;            // over B*NH*(S/16) = 2048 waves
  int mt = gw % (S_ / 16); gw /= (S_ / 16);
  int h  = gw % NH_;
  int b  = gw / NH_;
  int kv = h / (NH_ / NKV_);
  int m0 = mt * 16;
  int half = lane >> 4, l16 = lane & 15;

  // Preload Q fragments over HD (8 fragments of 16x32); Q pre-scaled by 1/16.
  const bf16_t* Qrow =
      Qb + (((size_t)(b * S_ + m0 + l16)) * NH_ + h) * HD_ + half * 8;
  v16bf qf[8];
#pragma unroll
  for (int kk = 0; kk < 8; ++kk)
    qf[kk] = load_frag_a(Qrow + kk * 32, Qrow + kk * 32 + 16);

  v8f oacc[16] = {};
  float mrun[8], lrun[8];
#pragma unroll
  for (int i = 0; i < 8; ++i) { mrun[i] = -1e30f; lrun[i] = 0.0f; }
  bf16_t* pl = &plds[wave][0];

  int nend = m0 + 16;                 // causal upper bound on keys
#if USE_ASYNC_LDS
  int nchunks = (nend >> 5) * 8;      // 8 k-chunks per 32-key tile
  int cbuf = 0;
  // lane L copies 64B of key row (n0 + L), k-dims kk*32..kk*32+31
  auto async_k = [&](int n0, int kk, int buf) {
    const bf16_t* g =
        Kb + (((size_t)(b * S_ + n0 + lane)) * NKV_ + kv) * HD_ + kk * 32;
    bf16_t* l = &klds[wave][buf][lane * 32];
#pragma unroll
    for (int seg = 0; seg < 4; ++seg)
      __builtin_amdgcn_global_load_async_to_lds_b128(
          (as1_v4i*)(g + seg * 8), (as3_v4i*)(l + seg * 8), 0, 0);
  };
  async_k(0, 0, 0);                   // prologue: chunk 0 -> buf 0
#endif

  for (int n0 = 0; n0 < nend; n0 += 32) {
    v8f s0 = {}, s1 = {};
#if !USE_ASYNC_LDS
    const bf16_t* Kb0 =
        Kb + (((size_t)(b * S_ + n0 + l16)) * NKV_ + kv) * HD_ + half * 8;
    const bf16_t* Kb1 = Kb0 + (size_t)16 * NKV_ * HD_;
#endif
#pragma unroll
    for (int kk = 0; kk < 8; ++kk) {
#if USE_ASYNC_LDS
      int c = (n0 >> 5) * 8 + kk;
      if (c + 1 < nchunks) {
        async_k(((c + 1) >> 3) << 5, (c + 1) & 7, cbuf ^ 1);
        __builtin_amdgcn_s_wait_asynccnt(4);  // cur chunk's 4 copies done
      } else {
        __builtin_amdgcn_s_wait_asynccnt(0);
      }
      v16bf b0 = load_frag_b(&klds[wave][cbuf][l16 * 32 + half * 8]);
      v16bf b1 = load_frag_b(&klds[wave][cbuf][(16 + l16) * 32 + half * 8]);
      cbuf ^= 1;
#else
      v16bf b0 = load_frag_b(Kb0 + kk * 32);
      v16bf b1 = load_frag_b(Kb1 + kk * 32);
#endif
      s0 = wmma_bf16(qf[kk], b0, s0);
      s1 = wmma_bf16(qf[kk], b1, s1);
    }
    // --- online softmax; P -> per-wave LDS tile (16x32 bf16) ---
#pragma unroll
    for (int i = 0; i < 8; ++i) {
      int r = i + half * 8;
      int qidx = m0 + r;
      int c0 = n0 + l16, c1 = n0 + 16 + l16;
      float e0 = (c0 <= qidx) ? s0[i] : -1e30f;
      float e1 = (c1 <= qidx) ? s1[i] : -1e30f;
      float tmax = fmaxf(e0, e1);
#pragma unroll
      for (int off = 1; off < 16; off <<= 1)
        tmax = fmaxf(tmax, __shfl_xor(tmax, off, 32));
      float mnew = fmaxf(mrun[i], tmax);
      float alpha = __expf(mrun[i] - mnew);
      float p0 = (c0 <= qidx) ? __expf(e0 - mnew) : 0.0f;
      float p1 = (c1 <= qidx) ? __expf(e1 - mnew) : 0.0f;
      float psum = p0 + p1;
#pragma unroll
      for (int off = 1; off < 16; off <<= 1) psum += __shfl_xor(psum, off, 32);
      lrun[i] = lrun[i] * alpha + psum;
      mrun[i] = mnew;
#pragma unroll
      for (int t = 0; t < 16; ++t) oacc[t][i] *= alpha;
      pl[r * 32 + l16]      = (bf16_t)p0;
      pl[r * 32 + 16 + l16] = (bf16_t)p1;
    }
    // --- P fragment (A, 16x32) from LDS; accumulate O += P*V ---
    v16bf pf = load_frag_a(pl + l16 * 32 + half * 8,
                           pl + l16 * 32 + 16 + half * 8);
    const bf16_t* Vrow =
        Vt + ((size_t)(b * NKV_ + kv) * HD_ + l16) * S_ + n0 + half * 8;
#pragma unroll
    for (int t = 0; t < 16; ++t) {
      v16bf vf = load_frag_b(Vrow + (size_t)t * 16 * S_);
      oacc[t] = wmma_bf16(pf, vf, oacc[t]);
    }
  }
  // --- normalize and store (bf16, (B,S,NH*HD)) ---
#pragma unroll
  for (int i = 0; i < 8; ++i) {
    float invl = 1.0f / lrun[i];
    int r = i + half * 8;
    bf16_t* orow = Ob + ((size_t)(b * S_ + m0 + r)) * (NH_ * HD_) + h * HD_;
#pragma unroll
    for (int t = 0; t < 16; ++t)
      orow[t * 16 + l16] = (bf16_t)(oacc[t][i] * invl);
  }
}

// ---------------------------------------------------------------------------
// Host-side orchestration
// ---------------------------------------------------------------------------
extern "C" void kernel_launch(void* const* d_in, const int* in_sizes, int n_in,
                              void* d_out, int out_size, void* d_ws,
                              size_t ws_size, hipStream_t stream) {
  const float* x    = (const float*)d_in[0];
  const float* cosb = (const float*)d_in[1];
  const float* sinb = (const float*)d_in[2];
  // d_in[3] = mask (causal -1e9): handled analytically, not read.
  const float* wq   = (const float*)d_in[4];
  const float* wk   = (const float*)d_in[5];
  const float* wv   = (const float*)d_in[6];
  const float* wo   = (const float*)d_in[7];
  const float* qnw  = (const float*)d_in[8];
  const float* knw  = (const float*)d_in[9];
  float* out = (float*)d_out;

  const int ROWS = B_ * S_;          // 4096
  const int QN   = NH_ * HD_;        // 2048
  const int KVN  = NKV_ * HD_;       // 512

  size_t off = 0;
  auto carve = [&](size_t bytes) {
    size_t o = off;
    off = (off + bytes + 255) & ~(size_t)255;
    return o;
  };
  char* ws = (char*)d_ws;
  bf16_t* xb   = (bf16_t*)(ws + carve((size_t)ROWS * HID_ * 2));
  bf16_t* wqT  = (bf16_t*)(ws + carve((size_t)QN * HID_ * 2));
  bf16_t* wkT  = (bf16_t*)(ws + carve((size_t)KVN * HID_ * 2));
  bf16_t* wvT  = (bf16_t*)(ws + carve((size_t)KVN * HID_ * 2));
  bf16_t* woT  = (bf16_t*)(ws + carve((size_t)HID_ * QN * 2));
  float*  Qf   = (float*)(ws + carve((size_t)ROWS * QN * 4));
  float*  Kf   = (float*)(ws + carve((size_t)ROWS * KVN * 4));
  float*  Vf   = (float*)(ws + carve((size_t)ROWS * KVN * 4));
  bf16_t* Qbb  = (bf16_t*)(ws + carve((size_t)ROWS * QN * 2));
  bf16_t* Kbb  = (bf16_t*)(ws + carve((size_t)ROWS * KVN * 2));
  bf16_t* Vt   = (bf16_t*)(ws + carve((size_t)ROWS * KVN * 2));
  bf16_t* Ob   = (bf16_t*)(ws + carve((size_t)ROWS * QN * 2));
  (void)ws_size; (void)in_sizes; (void)n_in; (void)out_size;

  // 1) precision conversion / layout prep
  cvt_f32_bf16_kernel<<<(ROWS * HID_ + 255) / 256, 256, 0, stream>>>(
      x, xb, ROWS * HID_);
  transpose_cvt_kernel<<<(HID_ * QN + 255) / 256, 256, 0, stream>>>(
      wq, wqT, HID_, QN);
  transpose_cvt_kernel<<<(HID_ * KVN + 255) / 256, 256, 0, stream>>>(
      wk, wkT, HID_, KVN);
  transpose_cvt_kernel<<<(HID_ * KVN + 255) / 256, 256, 0, stream>>>(
      wv, wvT, HID_, KVN);
  transpose_cvt_kernel<<<(QN * HID_ + 255) / 256, 256, 0, stream>>>(
      wo, woT, QN, HID_);

  // 2) QKV projections (WMMA GEMMs, 32x64 wave tiles)
  gemm_bf16_kernel<<<(ROWS / 32) * (QN / 64) / 8, 256, 0, stream>>>(
      xb, wqT, Qf, ROWS, QN, HID_);
  gemm_bf16_kernel<<<(ROWS / 32) * (KVN / 64) / 8, 256, 0, stream>>>(
      xb, wkT, Kf, ROWS, KVN, HID_);
  gemm_bf16_kernel<<<(ROWS / 32) * (KVN / 64) / 8, 256, 0, stream>>>(
      xb, wvT, Vf, ROWS, KVN, HID_);

  // 3) head RMS-norm + partial RoPE -> bf16 (Q pre-scaled by 1/sqrt(HD))
  rmsnorm_rope_kernel<<<(B_ * S_ * NH_) / 8, 256, 0, stream>>>(
      Qf, Qbb, qnw, cosb, sinb, NH_, 0.0625f);
  rmsnorm_rope_kernel<<<(B_ * S_ * NKV_) / 8, 256, 0, stream>>>(
      Kf, Kbb, knw, cosb, sinb, NKV_, 1.0f);

  // 4) V -> (B,NKV,HD,S) bf16 for contiguous P*V B-fragments
  v_transpose_kernel<<<(ROWS * KVN + 255) / 256, 256, 0, stream>>>(Vf, Vt);

  // 5) causal flash attention (WMMA + async K prefetch to LDS)
  attn_kernel<<<(B_ * NH_ * (S_ / 16)) / 8, 256, 0, stream>>>(Qbb, Kbb, Vt, Ob);

  // 6) output projection (WMMA GEMM), f32 result
  gemm_bf16_kernel<<<(ROWS / 32) * (HID_ / 64) / 8, 256, 0, stream>>>(
      Ob, woT, out, ROWS, HID_, QN);
}
